// TGNNPP_43868795961623
// MI455X (gfx1250) — matrix operand order repaired and checked by reading, not compiled, not run on hardware
//
#include <hip/hip_runtime.h>
#include <hip/hip_bf16.h>
#include <cstdint>
#include <cstddef>

typedef __attribute__((ext_vector_type(16))) __bf16 v16bf;
typedef __attribute__((ext_vector_type(8)))  float  v8f;
typedef unsigned int  u32x4 __attribute__((ext_vector_type(4)));
typedef int           i32x8 __attribute__((ext_vector_type(8)));
typedef int           i32x4 __attribute__((ext_vector_type(4)));
typedef unsigned short ushort_t;

union Frag { uint4 q2[2]; v16bf v; };   // 32 bytes: 16 bf16

__device__ __forceinline__ ushort_t f2bf(float f) {
  unsigned u = __float_as_uint(f);
  u += 0x7FFFu + ((u >> 16) & 1u);   // round-to-nearest-even
  return (ushort_t)(u >> 16);
}

// monotone float<->uint order mapping (for atomic max on floats)
__device__ __forceinline__ unsigned ford(float f) {
  int i = __float_as_int(f);
  return (unsigned)(i ^ ((i >> 31) | 0x80000000));
}
__device__ __forceinline__ float funord(unsigned m) {
  int i = (m & 0x80000000u) ? (int)(m ^ 0x80000000u) : (int)(~m);
  return __int_as_float(i);
}

// ---------------------------------------------------------------------------
// precision-conversion pre-passes
// ---------------------------------------------------------------------------
__global__ void cvt_f32_bf16(const float* __restrict__ in, ushort_t* __restrict__ out,
                             size_t n) {
  size_t i = (size_t)blockIdx.x * blockDim.x + threadIdx.x;
  if (i < n) out[i] = f2bf(in[i]);
}

// weights: (K,N) f32 row-major -> (N,K) bf16 row-major
__global__ void cvt_transpose_bf16(const float* __restrict__ in, ushort_t* __restrict__ out,
                                   int K, int N) {
  int i = blockIdx.x * blockDim.x + threadIdx.x;
  if (i >= K * N) return;
  int k = i / N, n = i % N;
  out[(size_t)n * K + k] = f2bf(in[i]);
}

// ---------------------------------------------------------------------------
// C = act(A @ B + bias).  A:(M,K) bf16 row-major, Bt:(N,K) bf16 row-major.
// Grid: (ceil(M/16/8), N/(16*NB)); 8 waves/block share one (16*NB)-column B
// panel staged into LDS by the Tensor Data Mover. Each wave computes a
// 16 x (16*NB) strip: one A-fragment load feeds NB WMMAs per k-step.
// ---------------------------------------------------------------------------
template <int NB>
__global__ void gemm_bf16_wmma(const ushort_t* __restrict__ A,
                               const ushort_t* __restrict__ Bt,
                               const float* __restrict__ bias,
                               float* __restrict__ Cf,        // optional f32 out
                               ushort_t* __restrict__ Cb,     // optional bf16 out
                               int M, int N, int K, int relu) {
  extern __shared__ ushort_t sB[];           // (16*NB) rows x K bf16 panel
  const int lane  = threadIdx.x & 31;
  const int wid   = threadIdx.x >> 5;
  const int mtiles = M >> 4;
  const int prows = 16 * NB;                 // panel rows (columns of C)

  const size_t panel = (size_t)blockIdx.y * prows * K;

#if defined(__gfx1250__) && __has_builtin(__builtin_amdgcn_tensor_load_to_lds) && \
    __has_builtin(__builtin_amdgcn_s_wait_tensorcnt)
  if (wid == 0) {
    // Tensor DMA descriptor (D#): 2-D tile, prows rows x K elements, 2B data.
    unsigned long long ga = (unsigned long long)(const void*)(Bt + panel);
    unsigned lds = (unsigned)(size_t)(void*)&sB[0];
    u32x4 g0;
    g0.x = 1u;                                              // count=1, user desc
    g0.y = lds;                                             // lds_addr
    g0.z = (unsigned)(ga & 0xFFFFFFFFu);                    // global_addr[31:0]
    g0.w = (unsigned)((ga >> 32) & 0x1FFFFFFu) | (2u << 30);// addr[56:32] | type=2
    i32x8 g1;
    g1[0] = 0x00010000;                       // workgroup_mask=0, data_size=2B
    g1[1] = (int)(((unsigned)K & 0xFFFFu) << 16);             // tensor_dim0[15:0]
    g1[2] = (int)(((unsigned)K >> 16) | ((unsigned)prows << 16)); // td0 hi | td1
    g1[3] = (int)((unsigned)K << 16);         // td1 hi=0 | tile_dim0=K
    g1[4] = prows;                            // tile_dim1=prows, tile_dim2=0
    g1[5] = K;                                // tensor_dim0_stride[31:0]=K
    g1[6] = 0; g1[7] = 0;
    i32x4 z4 = {0, 0, 0, 0};
#if __clang_major__ >= 23
    i32x8 z8 = {0, 0, 0, 0, 0, 0, 0, 0};
    __builtin_amdgcn_tensor_load_to_lds(g0, g1, z4, z4, z8, 0);
#else
    __builtin_amdgcn_tensor_load_to_lds(g0, g1, z4, z4, 0);
#endif
    __builtin_amdgcn_s_wait_tensorcnt(0);
  }
  __syncthreads();
#else
  for (int i = threadIdx.x; i < prows * (K >> 3); i += blockDim.x)
    ((uint4*)sB)[i] = ((const uint4*)(Bt + panel))[i];
  __syncthreads();
#endif

  int mt = blockIdx.x * 8 + wid;
  if (mt >= mtiles) return;                 // wave-uniform: EXEC all-ones for WMMA

  const int lrow = lane & 15;               // A row / B col / D col in tile
  const int lhi  = lane >> 4;

  const ushort_t* Ap = A + (size_t)(mt * 16 + lrow) * K + lhi * 8;
  const ushort_t* Bp = sB + (size_t)lrow * K + lhi * 8;

  v8f acc[NB];
#pragma unroll
  for (int s = 0; s < NB; ++s) acc[s] = (v8f){0.f,0.f,0.f,0.f,0.f,0.f,0.f,0.f};

  for (int k0 = 0; k0 < K; k0 += 32) {
    Frag a;
    a.q2[0] = *(const uint4*)(Ap + k0);        // k = k0 + lhi*8 + (0..7)
    a.q2[1] = *(const uint4*)(Ap + k0 + 16);   // k = k0 + 16 + lhi*8 + (0..7)
#pragma unroll
    for (int s = 0; s < NB; ++s) {
      Frag b;
      const ushort_t* bp = Bp + (size_t)s * 16 * K + k0;
      b.q2[0] = *(const uint4*)(bp);           // k = k0 + lhi*8 + (0..7)
      b.q2[1] = *(const uint4*)(bp + 16);      // k = k0 + 16 + lhi*8 + (0..7)
      acc[s] = __builtin_amdgcn_wmma_f32_16x16x32_bf16(
          false, a.v, false, b.v, (short)0, acc[s], false, false);
    }
  }

#pragma unroll
  for (int s = 0; s < NB; ++s) {
    int col = (blockIdx.y * NB + s) * 16 + lrow;
    float bs = bias ? bias[col] : 0.f;
#pragma unroll
    for (int r = 0; r < 8; ++r) {
      int row = mt * 16 + (lhi << 3) + r;
      float v = acc[s][r] + bs;
      if (relu) v = fmaxf(v, 0.f);
      if (Cf) Cf[(size_t)row * N + col] = v;
      if (Cb) Cb[(size_t)row * N + col] = f2bf(v);
    }
  }
}

// ---------------------------------------------------------------------------
// GAT pieces
// ---------------------------------------------------------------------------
__global__ void zero32_kernel(unsigned* p, size_t n) {
  size_t i = (size_t)blockIdx.x * blockDim.x + threadIdx.x;
  if (i < n) p[i] = 0u;
}

__global__ void build_edges(const int* __restrict__ ei, int* __restrict__ src,
                            int* __restrict__ dst, int B, int E, int N, int M) {
  int i = blockIdx.x * blockDim.x + threadIdx.x;
  int tot = B * E + M;
  if (i >= tot) return;
  if (i < B * E) {
    int b = i / E, e = i % E;
    src[i] = ei[(b * 2 + 0) * E + e] + b * N;
    dst[i] = ei[(b * 2 + 1) * E + e] + b * N;
  } else {
    int m = i - B * E;
    src[i] = m; dst[i] = m;                  // self loops
  }
}

__global__ void gat_attn_coef(const float* __restrict__ h,
                              const float* __restrict__ att_src,
                              const float* __restrict__ att_dst,
                              float* __restrict__ a_src, float* __restrict__ a_dst,
                              int rows) {
  int i = blockIdx.x * blockDim.x + threadIdx.x;
  if (i >= rows * 2) return;
  int row = i >> 1, hd = i & 1;
  const float* hp = h + (size_t)row * 128 + hd * 64;
  const float* as = att_src + hd * 64;
  const float* ad = att_dst + hd * 64;
  float s = 0.f, d = 0.f;
  for (int c = 0; c < 64; ++c) { float v = hp[c]; s += v * as[c]; d += v * ad[c]; }
  a_src[row * 2 + hd] = s;
  a_dst[row * 2 + hd] = d;
}

__global__ void gat_edge_logits(const float* __restrict__ a_src,
                                const float* __restrict__ a_dst,
                                const int* __restrict__ src, const int* __restrict__ dst,
                                float* __restrict__ evals, unsigned* __restrict__ mx,
                                int NE, int N, int W, int M) {
  int i = blockIdx.x * blockDim.x + threadIdx.x;
  if (i >= NE * W) return;
  int t = i / NE, e = i % NE;
  int s = src[e], d = dst[e];
  int rs = (s / N) * W * N + t * N + (s % N);
  int rd = (d / N) * W * N + t * N + (d % N);
#pragma unroll
  for (int hd = 0; hd < 2; ++hd) {
    float v = a_src[rs * 2 + hd] + a_dst[rd * 2 + hd];
    v = v > 0.f ? v : 0.2f * v;
    evals[((size_t)t * NE + e) * 2 + hd] = v;
    atomicMax(&mx[((size_t)t * M + d) * 2 + hd], ford(v));
  }
}

__global__ void gat_edge_exp(const float* __restrict__ evals,
                             const unsigned* __restrict__ mx,
                             const int* __restrict__ dst,
                             float* __restrict__ exv, float* __restrict__ den,
                             int NE, int W, int M) {
  int i = blockIdx.x * blockDim.x + threadIdx.x;
  if (i >= NE * W) return;
  int t = i / NE, e = i % NE;
  int d = dst[e];
#pragma unroll
  for (int hd = 0; hd < 2; ++hd) {
    size_t idx = ((size_t)t * NE + e) * 2 + hd;
    float ev = __expf(evals[idx] - funord(mx[((size_t)t * M + d) * 2 + hd]));
    exv[idx] = ev;
    atomicAdd(&den[((size_t)t * M + d) * 2 + hd], ev);
  }
}

__global__ void gat_scatter(const float* __restrict__ h,
                            const int* __restrict__ src, const int* __restrict__ dst,
                            const float* __restrict__ exv, const float* __restrict__ den,
                            float* __restrict__ gout,
                            int NE, int N, int W, int M) {
  int wave = blockIdx.x * 8 + (threadIdx.x >> 5);
  int lane = threadIdx.x & 31;
  if (wave >= NE * W) return;
  int t = wave / NE, e = wave % NE;
  int s = src[e], d = dst[e];
  int rs = (s / N) * W * N + t * N + (s % N);
  int c0 = lane * 4;
  int hd = c0 >> 6;
  float alpha = exv[((size_t)t * NE + e) * 2 + hd] /
                den[((size_t)t * M + d) * 2 + hd];
  const float4 hv = *(const float4*)(h + (size_t)rs * 128 + c0);
  float* gp = gout + ((size_t)t * M + d) * 128 + c0;
  atomicAdd(gp + 0, alpha * hv.x);
  atomicAdd(gp + 1, alpha * hv.y);
  atomicAdd(gp + 2, alpha * hv.z);
  atomicAdd(gp + 3, alpha * hv.w);
}

__global__ void temporal_mean(const float* __restrict__ gout,
                              const float* __restrict__ gat_bias,
                              float* __restrict__ y, int B, int W, int N, int M) {
  int b = blockIdx.x / W, t = blockIdx.x % W;
  int c = threadIdx.x;
  const float* gp = gout + ((size_t)t * M + b * N) * 128 + c;
  float s = 0.f;
  for (int n = 0; n < N; ++n) s += gp[(size_t)n * 128];
  y[((size_t)b * W + t) * 128 + c] = s / (float)N + gat_bias[c];
}

// ---------------------------------------------------------------------------
// Transformer pieces (tiny: 240 tokens x 128)
// ---------------------------------------------------------------------------
__global__ void mha_kernel(const float* __restrict__ qkv, float* __restrict__ o,
                           int B, int W, int d, int nh) {
  int b = blockIdx.x / nh, hh = blockIdx.x % nh;
  int hd = d / nh;                           // 16
  int q = threadIdx.x;
  if (q >= W) return;
  const float* qp = qkv + (size_t)(b * W + q) * 3 * d + hh * hd;
  float qr[16];
#pragma unroll
  for (int c = 0; c < 16; ++c) qr[c] = qp[c];
  float sc[32];
  float mxv = -1e30f;
  for (int k = 0; k < W; ++k) {
    const float* kp = qkv + (size_t)(b * W + k) * 3 * d + d + hh * hd;
    float dot = 0.f;
#pragma unroll
    for (int c = 0; c < 16; ++c) dot += qr[c] * kp[c];
    dot *= 0.25f;                            // 1/sqrt(16)
    sc[k] = dot;
    mxv = fmaxf(mxv, dot);
  }
  float dn = 0.f;
  for (int k = 0; k < W; ++k) { sc[k] = __expf(sc[k] - mxv); dn += sc[k]; }
  float inv = 1.f / dn;
  float ov[16];
#pragma unroll
  for (int c = 0; c < 16; ++c) ov[c] = 0.f;
  for (int k = 0; k < W; ++k) {
    const float* vp = qkv + (size_t)(b * W + k) * 3 * d + 2 * d + hh * hd;
    float a = sc[k] * inv;
#pragma unroll
    for (int c = 0; c < 16; ++c) ov[c] += a * vp[c];
  }
  float* op = o + (size_t)(b * W + q) * d + hh * hd;
#pragma unroll
  for (int c = 0; c < 16; ++c) op[c] = ov[c];
}

__global__ void add_ln(float* __restrict__ y, const float* __restrict__ r,
                       const float* __restrict__ g, const float* __restrict__ bta,
                       int d) {
  int tkn = blockIdx.x, c = threadIdx.x;
  __shared__ float red[128];
  float v = y[(size_t)tkn * d + c] + r[(size_t)tkn * d + c];
  red[c] = v;
  __syncthreads();
  for (int s2 = 64; s2 > 0; s2 >>= 1) {
    if (c < s2) red[c] += red[c + s2];
    __syncthreads();
  }
  float mu = red[0] / (float)d;
  __syncthreads();
  float dv = v - mu;
  red[c] = dv * dv;
  __syncthreads();
  for (int s2 = 64; s2 > 0; s2 >>= 1) {
    if (c < s2) red[c] += red[c + s2];
    __syncthreads();
  }
  float var = red[0] / (float)d;
  y[(size_t)tkn * d + c] = dv * rsqrtf(var + 1e-5f) * g[c] + bta[c];
}

__global__ void pred_head(const float* __restrict__ y,
                          const float* __restrict__ w1, const float* __restrict__ b1,
                          const float* __restrict__ w2, const float* __restrict__ b2,
                          float* __restrict__ out, int B, int W, int d) {
  int b = blockIdx.x, j = threadIdx.x;       // 64 threads
  const float* f = y + ((size_t)b * W + (W - 1)) * d;
  float s = b1[j];
  for (int c = 0; c < d; ++c) s += f[c] * w1[c * 64 + j];
  s = fmaxf(s, 0.f);
  __shared__ float red[64];
  red[j] = s * w2[j];
  __syncthreads();
  for (int st = 32; st > 0; st >>= 1) {
    if (j < st) red[j] += red[j + st];
    __syncthreads();
  }
  if (j == 0) out[b] = red[0] + b2[0];
}

// ---------------------------------------------------------------------------
extern "C" void kernel_launch(void* const* d_in, const int* in_sizes, int n_in,
                              void* d_out, int out_size, void* d_ws, size_t ws_size,
                              hipStream_t stream) {
  const float* node_feats = (const float*)d_in[0];
  const int*   edge_idx   = (const int*)d_in[1];
  const float* fe_w1 = (const float*)d_in[2];
  const float* fe_b1 = (const float*)d_in[3];
  const float* fe_w2 = (const float*)d_in[4];
  const float* fe_b2 = (const float*)d_in[5];
  const float* gat_w = (const float*)d_in[6];
  const float* att_src = (const float*)d_in[7];
  const float* att_dst = (const float*)d_in[8];
  const float* gat_bias = (const float*)d_in[9];
  const float* t_wqkv = (const float*)d_in[10];
  const float* t_bqkv = (const float*)d_in[11];
  const float* t_wo  = (const float*)d_in[12];
  const float* t_bo  = (const float*)d_in[13];
  const float* t_ln1g = (const float*)d_in[14];
  const float* t_ln1b = (const float*)d_in[15];
  const float* t_w1 = (const float*)d_in[16];
  const float* t_b1 = (const float*)d_in[17];
  const float* t_w2 = (const float*)d_in[18];
  const float* t_b2 = (const float*)d_in[19];
  const float* t_ln2g = (const float*)d_in[20];
  const float* t_ln2b = (const float*)d_in[21];
  const float* ph_w1 = (const float*)d_in[22];
  const float* ph_b1 = (const float*)d_in[23];
  const float* ph_w2 = (const float*)d_in[24];
  const float* ph_b2 = (const float*)d_in[25];
  float* out = (float*)d_out;

  constexpr int B = 8, W = 30, N = 500, E = 8000, D = 128, L = 4, DFF = 2048, NH = 8;
  constexpr int M = B * N;        // 4000
  constexpr int ROWS = B * W * N; // 120000
  constexpr int NE = B * E + M;   // 68000
  const int T = B * W;            // 240 tokens

  char* ws = (char*)d_ws;
  ushort_t* Xb      = (ushort_t*)(ws + 0);            // 120000*64 bf16 (15.4MB)
  ushort_t* hid_b   = (ushort_t*)(ws + 16000000ull);  // 120000*256 bf16 (61.4MB)
  ushort_t* emb_b   = (ushort_t*)(ws + 78000000ull);  // 120000*128 bf16 (30.7MB)
  float*    h       = (float*)(ws + 109000000ull);    // 120000*128 f32  (61.4MB)
  float*    gout    = (float*)(ws + 171000000ull);    // 30*4000*128 f32 (61.4MB)
  // bf16 weights (transposed to N x K)
  ushort_t* w1t     = (ushort_t*)(ws + 233000000ull); // 256*64
  ushort_t* w2t     = (ushort_t*)(ws + 233100000ull); // 128*256
  ushort_t* gwt     = (ushort_t*)(ws + 233200000ull); // 128*128
  ushort_t* wlt     = (ushort_t*)(ws + 234000000ull); // per-layer block, 1.18MB each
  const size_t LQKV = 0, LO = 98304 / 2, LW1 = (98304 + 32768) / 2,
               LW2 = (98304 + 32768 + 524288) / 2, LSTRIDE = 1179648 / 2; // ushort idx
  // GAT scratch
  float*    evals   = (float*)(ws + 240000000ull);    // 30*68000*2
  float*    exv     = (float*)(ws + 257000000ull);
  unsigned* mx      = (unsigned*)(ws + 274000000ull);
  float*    den     = (float*)(ws + 275000000ull);
  float*    asrc    = (float*)(ws + 276000000ull);
  float*    adst    = (float*)(ws + 277000000ull);
  int*      srcA    = (int*)(ws + 278000000ull);
  int*      dstA    = (int*)(ws + 278400000ull);
  // transformer scratch
  float*    y       = (float*)(ws + 279000000ull);    // 240*128
  float*    qkv     = (float*)(ws + 279200000ull);    // 240*384
  float*    atto    = (float*)(ws + 279600000ull);    // 240*128
  float*    o2      = (float*)(ws + 279800000ull);    // 240*128
  ushort_t* ffb     = (ushort_t*)(ws + 280000000ull); // 240*2048 bf16
  ushort_t* ybf     = (ushort_t*)(ws + 281200000ull); // 240*128 bf16
  ushort_t* attob   = (ushort_t*)(ws + 281400000ull); // 240*128 bf16

  // NB=4: 64-col B panels, 4 WMMAs per A-fragment load (A HBM traffic / 4)
  auto gemm4 = [&](const ushort_t* A, const ushort_t* Bt, const float* bias,
                   float* Cf, ushort_t* Cb, int Mi, int Ni, int Ki, int relu) {
    dim3 grid((Mi / 16 + 7) / 8, Ni / 64);
    gemm_bf16_wmma<4><<<grid, 256, (size_t)64 * Ki * 2, stream>>>(
        A, Bt, bias, Cf, Cb, Mi, Ni, Ki, relu);
  };
  // NB=1: for huge-K GEMMs where a 64-col panel would exceed LDS
  auto gemm1 = [&](const ushort_t* A, const ushort_t* Bt, const float* bias,
                   float* Cf, ushort_t* Cb, int Mi, int Ni, int Ki, int relu) {
    dim3 grid((Mi / 16 + 7) / 8, Ni / 16);
    gemm_bf16_wmma<1><<<grid, 256, (size_t)16 * Ki * 2, stream>>>(
        A, Bt, bias, Cf, Cb, Mi, Ni, Ki, relu);
  };
  auto cvt = [&](const float* in, ushort_t* o_, size_t n) {
    cvt_f32_bf16<<<(unsigned)((n + 255) / 256), 256, 0, stream>>>(in, o_, n);
  };
  auto cvtT = [&](const float* in, ushort_t* o_, int K, int Nn) {
    cvt_transpose_bf16<<<(K * Nn + 255) / 256, 256, 0, stream>>>(in, o_, K, Nn);
  };

  // ---- convert inputs/weights to bf16 (weights transposed) ----
  cvt(node_feats, Xb, (size_t)ROWS * 64);
  cvtT(fe_w1, w1t, 64, 256);
  cvtT(fe_w2, w2t, 256, 128);
  cvtT(gat_w, gwt, 128, 128);
  for (int l = 0; l < L; ++l) {
    cvtT(t_wqkv + (size_t)l * D * 3 * D, wlt + l * LSTRIDE + LQKV, D, 3 * D);
    cvtT(t_wo  + (size_t)l * D * D,     wlt + l * LSTRIDE + LO,   D, D);
    cvtT(t_w1  + (size_t)l * D * DFF,   wlt + l * LSTRIDE + LW1,  D, DFF);
    cvtT(t_w2  + (size_t)l * DFF * D,   wlt + l * LSTRIDE + LW2,  DFF, D);
  }

  // ---- feature encoder: emb = relu(X@W1+b1)@W2+b2 (rows ordered (b,w,n)) ----
  gemm4(Xb, w1t, fe_b1, nullptr, hid_b, ROWS, 256, 64, 1);
  gemm4(hid_b, w2t, fe_b2, nullptr, emb_b, ROWS, 128, 256, 0);

  // ---- GAT projection: h = emb @ gat_w (f32 for the scatter math) ----
  gemm4(emb_b, gwt, nullptr, h, nullptr, ROWS, 128, 128, 0);

  // ---- GAT attention ----
  build_edges<<<(NE + 255) / 256, 256, 0, stream>>>(edge_idx, srcA, dstA, B, E, N, M);
  gat_attn_coef<<<(ROWS * 2 + 255) / 256, 256, 0, stream>>>(h, att_src, att_dst,
                                                            asrc, adst, ROWS);
  zero32_kernel<<<(W * M * 2 + 255) / 256, 256, 0, stream>>>(mx, (size_t)W * M * 2);
  zero32_kernel<<<(W * M * 2 + 255) / 256, 256, 0, stream>>>((unsigned*)den,
                                                             (size_t)W * M * 2);
  zero32_kernel<<<((size_t)W * M * 128 + 255) / 256, 256, 0, stream>>>(
      (unsigned*)gout, (size_t)W * M * 128);

  int ethreads = NE * W;
  gat_edge_logits<<<(ethreads + 255) / 256, 256, 0, stream>>>(asrc, adst, srcA, dstA,
                                                              evals, mx, NE, N, W, M);
  gat_edge_exp<<<(ethreads + 255) / 256, 256, 0, stream>>>(evals, mx, dstA, exv, den,
                                                           NE, W, M);
  gat_scatter<<<(ethreads + 7) / 8, 256, 0, stream>>>(h, srcA, dstA, exv, den, gout,
                                                      NE, N, W, M);

  // ---- temporal mean over nodes (+gat_bias) -> y (B,W,128) ----
  temporal_mean<<<B * W, 128, 0, stream>>>(gout, gat_bias, y, B, W, N, M);

  // ---- transformer: 4 layers on 240 tokens ----
  for (int l = 0; l < L; ++l) {
    cvt(y, ybf, (size_t)T * D);
    gemm4(ybf, wlt + l * LSTRIDE + LQKV, t_bqkv + (size_t)l * 3 * D, qkv, nullptr,
          T, 3 * D, D, 0);
    mha_kernel<<<B * NH, 32, 0, stream>>>(qkv, atto, B, W, D, NH);
    cvt(atto, attob, (size_t)T * D);
    gemm4(attob, wlt + l * LSTRIDE + LO, t_bo + (size_t)l * D, o2, nullptr, T, D, D, 0);
    add_ln<<<T, 128, 0, stream>>>(y, o2, t_ln1g + l * D, t_ln1b + l * D, D);
    cvt(y, ybf, (size_t)T * D);
    gemm4(ybf, wlt + l * LSTRIDE + LW1, t_b1 + (size_t)l * DFF, nullptr, ffb,
          T, DFF, D, 1);
    gemm1(ffb, wlt + l * LSTRIDE + LW2, t_b2 + (size_t)l * D, o2, nullptr,
          T, D, DFF, 0);                    // K=2048: 16-col panel (64KB LDS)
    add_ln<<<T, 128, 0, stream>>>(y, o2, t_ln2g + l * D, t_ln2b + l * D, D);
  }

  // ---- prediction head ----
  pred_head<<<B, 64, 0, stream>>>(y, ph_w1, ph_b1, ph_w2, ph_b2, out, B, W, D);
}
